// ATMLayer_661424963817
// MI455X (gfx1250) — matrix operand order repaired and checked by reading, not compiled, not run on hardware
//
#include <hip/hip_runtime.h>
#include <hip/hip_fp16.h>

// MI455X / gfx1250 implementation of the ATM layer.
// B=16, H=64, W=64, C=256.  M = B*H*W = 65536 pixel-rows, K = N = 256.
// All three big GEMMs go through v_wmma_f32_16x16x32_f16 (wave32).
// Inter-kernel scratch tensors carried at f16 to halve HBM round-trip traffic
// (workload is memory-bound: ~26 GFLOP vs ~0.6 GB @ 23.3 TB/s).

typedef __attribute__((ext_vector_type(16))) _Float16 v16h;
typedef __attribute__((ext_vector_type(8)))  float    v8f;

// ---------------------------------------------------------------------------
// WMMA fragment index helpers (CDNA5 ISA 7.12.2, 16-bit A 16x32 / B 32x16)
//   A (16x32, MxK): lane = (m&15) + 16*khalf (khalf=(kk>>3)&1),
//                   element e = 8*(kk>>4) + (kk&7)   (kk = k mod 32)
// Fragments stored contiguously: [(mtile*8 + kstep)*32 + lane]*16 + e
// so each lane reads its 16 f16 (32 bytes) with contiguous b128 loads.
// ---------------------------------------------------------------------------
__device__ __forceinline__ size_t fragA_idx(int m, int k) {
  int mtile = m >> 4;
  int kstep = k >> 5;
  int kk    = k & 31;
  int khalf = (kk >> 3) & 1;
  int lane  = (m & 15) | (khalf << 4);
  int e     = ((kk >> 4) << 3) | (kk & 7);
  return ((((size_t)mtile * 8 + kstep) * 32 + lane) << 4) + e;
}

// ---------------------------------------------------------------------------
// Kernel 0: zero the per-batch reduction buffer (ws is poisoned once, and we
// accumulate into it with atomics every launch).
// ---------------------------------------------------------------------------
__global__ void k_zero(float* __restrict__ p, int n) {
  int i = blockIdx.x * blockDim.x + threadIdx.x;
  if (i < n) p[i] = 0.f;
}

// ---------------------------------------------------------------------------
// Kernel 1: pre-swizzle the three 256x256 weight matrices (row-major [o,c])
// into f16 WMMA B-fragment order.  B[k][n] = weight[n][k].
// B fragment: lane = n + 16*khalf, element e -> k = kstep*32 + (e>>3)*16 + khalf*8 + (e&7).
// Only 3 x 128 KB: lives in L2 for the whole dispatch.
// ---------------------------------------------------------------------------
__global__ void k_prep_b(const float* __restrict__ w0, const float* __restrict__ w1,
                         const float* __restrict__ w2,
                         _Float16* __restrict__ b0, _Float16* __restrict__ b1,
                         _Float16* __restrict__ b2) {
  int mat = blockIdx.x >> 8;                         // 0..2
  int d   = ((blockIdx.x & 255) << 8) + threadIdx.x; // 0..65535
  const float* src = (mat == 0) ? w0 : (mat == 1) ? w1 : w2;
  _Float16*    dst = (mat == 0) ? b0 : (mat == 1) ? b1 : b2;
  int lane  = (d >> 4) & 31;
  int e     = d & 15;
  int frag  = d >> 9;
  int kstep = frag & 7;
  int ntile = frag >> 3;
  int n     = lane & 15;
  int khalf = lane >> 4;
  int k     = kstep * 32 + ((e >> 3) << 4) + khalf * 8 + (e & 7);
  dst[d] = (_Float16)src[(ntile * 16 + n) * 256 + k];
}

// ---------------------------------------------------------------------------
// Kernel 2: deform sampling (1-D lerp in x for the w-branch, 1-D lerp in y
// for the h-branch; dy/dx == 0 makes the other axis exact) + 5x5 depthwise
// conv.  Writes sampled branches straight into WMMA A-fragment layout (f16),
// conv branch as f16 row-major.
// grid = M/16 blocks, 256 threads (thread = channel), 16 pixels per block.
// ---------------------------------------------------------------------------
__global__ void k_sample(const float* __restrict__ x, const float* __restrict__ off,
                         const float* __restrict__ lw, const float* __restrict__ lb,
                         _Float16* __restrict__ Aw, _Float16* __restrict__ Ah,
                         _Float16* __restrict__ Cb) {
  const int ch = threadIdx.x;
  const int p0 = blockIdx.x << 4;
  float lwv[25];
#pragma unroll
  for (int t = 0; t < 25; ++t) lwv[t] = lw[ch * 25 + t];
  const float lbv = lb[ch];

  for (int i = 0; i < 16; ++i) {
    const int m   = p0 + i;
    const int b   = m >> 12;       // H*W = 4096
    const int rem = m & 4095;
    const int y   = rem >> 6;
    const int xx  = rem & 63;
    const float* xb = x + (size_t)b * 4096 * 256;   // NHWC batch base

    // w-branch: x-offset only (offset channels [0,C))
    float ox  = off[(((size_t)(b * 512 + ch) * 64 + y) * 64) + xx];
    float xs  = (float)xx + ox;
    float x0f = floorf(xs);
    float wx  = xs - x0f;
    int   x0  = (int)x0f;
    float v0  = (x0 >= 0 && x0 < 64)         ? xb[(y * 64 + x0) * 256 + ch]     : 0.f;
    float v1  = (x0 + 1 >= 0 && x0 + 1 < 64) ? xb[(y * 64 + x0 + 1) * 256 + ch] : 0.f;
    float sw  = (1.f - wx) * v0 + wx * v1;

    // h-branch: y-offset only (offset channels [C,2C))
    float oy  = off[(((size_t)(b * 512 + 256 + ch) * 64 + y) * 64) + xx];
    float ysv = (float)y + oy;
    float y0f = floorf(ysv);
    float wy  = ysv - y0f;
    int   y0  = (int)y0f;
    float u0  = (y0 >= 0 && y0 < 64)         ? xb[(y0 * 64 + xx) * 256 + ch]       : 0.f;
    float u1  = (y0 + 1 >= 0 && y0 + 1 < 64) ? xb[((y0 + 1) * 64 + xx) * 256 + ch] : 0.f;
    float sh  = (1.f - wy) * u0 + wy * u1;

    // depthwise 5x5, zero pad, + local_bias
    float acc = lbv;
#pragma unroll
    for (int dy = -2; dy <= 2; ++dy) {
      int yy = y + dy;
      if (yy < 0 || yy >= 64) continue;
#pragma unroll
      for (int dx = -2; dx <= 2; ++dx) {
        int xc = xx + dx;
        if (xc < 0 || xc >= 64) continue;
        acc += xb[(yy * 64 + xc) * 256 + ch] * lwv[(dy + 2) * 5 + (dx + 2)];
      }
    }

    Aw[fragA_idx(m, ch)] = (_Float16)sw;
    Ah[fragA_idx(m, ch)] = (_Float16)sh;
    Cb[(size_t)m * 256 + ch] = (_Float16)acc;
  }
}

// ---------------------------------------------------------------------------
// Kernel 3: fused dual WMMA GEMM:  w = Sw*Ww^T + bw, h = Sh*Wh^T + bh, plus
// per-batch partial sum of (w+h+c) for the gating mean (f32 in LDS/atomics).
// Block = 256 threads = 8 waves; block owns one 16-row M-tile, full N=256;
// each wave does 2 N-tiles for both branches -> 32 v_wmma per wave.
// Branch outputs stored f16 row-major.
// ---------------------------------------------------------------------------
__global__ void k_gemm_wh(const _Float16* __restrict__ Aw, const _Float16* __restrict__ Ah,
                          const _Float16* __restrict__ Bw, const _Float16* __restrict__ Bh,
                          const float* __restrict__ wbias, const float* __restrict__ hbias,
                          const _Float16* __restrict__ Cb,
                          _Float16* __restrict__ Wb, _Float16* __restrict__ Hb,
                          float* __restrict__ psum) {
  __shared__ float sred[256];
  const int mtile = blockIdx.x;
  const int wave  = threadIdx.x >> 5;
  const int lane  = threadIdx.x & 31;
  sred[threadIdx.x] = 0.f;
  __syncthreads();

  const int nt0 = wave * 2;
  v8f accw0 = {}, accw1 = {}, acch0 = {}, acch1 = {};
  const size_t abase = (size_t)mtile * (8 * 32 * 16);

#pragma unroll
  for (int ks = 0; ks < 8; ++ks) {
    const size_t lao = abase + ((size_t)(ks * 32 + lane) << 4);
    v16h aw = *(const v16h*)(Aw + lao);
    v16h ah = *(const v16h*)(Ah + lao);
    const size_t b0o = ((size_t)((nt0 + 0) * 8 + ks) * 32 + lane) << 4;
    const size_t b1o = ((size_t)((nt0 + 1) * 8 + ks) * 32 + lane) << 4;
    v16h bw0 = *(const v16h*)(Bw + b0o);
    v16h bw1 = *(const v16h*)(Bw + b1o);
    v16h bh0 = *(const v16h*)(Bh + b0o);
    v16h bh1 = *(const v16h*)(Bh + b1o);
    accw0 = __builtin_amdgcn_wmma_f32_16x16x32_f16(false, aw, false, bw0, (short)0, accw0, false, false);
    accw1 = __builtin_amdgcn_wmma_f32_16x16x32_f16(false, aw, false, bw1, (short)0, accw1, false, false);
    acch0 = __builtin_amdgcn_wmma_f32_16x16x32_f16(false, ah, false, bh0, (short)0, acch0, false, false);
    acch1 = __builtin_amdgcn_wmma_f32_16x16x32_f16(false, ah, false, bh1, (short)0, acch1, false, false);
  }

  // C/D layout: VGPR r -> row M = r + 8*(lane>>4), col N = lane&15.
  const int mrow0 = (lane >> 4) << 3;
  const int ncol  = lane & 15;
#pragma unroll
  for (int t = 0; t < 2; ++t) {
    v8f aw = t ? accw1 : accw0;
    v8f ah = t ? acch1 : acch0;
    const int n   = (nt0 + t) * 16 + ncol;
    const float wbv = wbias[n], hbv = hbias[n];
    float s = 0.f;
#pragma unroll
    for (int r = 0; r < 8; ++r) {
      const int mg = mtile * 16 + mrow0 + r;
      const float wv = aw[r] + wbv;
      const float hv = ah[r] + hbv;
      const size_t o = (size_t)mg * 256 + n;
      Wb[o] = (_Float16)wv;
      Hb[o] = (_Float16)hv;
      s += wv + hv + (float)Cb[o];
    }
    atomicAdd(&sred[n], s);   // ds_add_f32
  }
  __syncthreads();
  const int bidx = mtile >> 8;  // 256 M-tiles per batch image
  atomicAdd(&psum[bidx * 256 + threadIdx.x], sred[threadIdx.x]);
}

// ---------------------------------------------------------------------------
// Kernel 4: gating MLP.  mean -> fc1 -> exact GELU -> fc2 -> 3-way softmax.
// Tiny (16 x 256): one block, serial over batch.
// ---------------------------------------------------------------------------
__global__ void k_gate(const float* __restrict__ psum,
                       const float* __restrict__ f1w, const float* __restrict__ f1b,
                       const float* __restrict__ f2w, const float* __restrict__ f2b,
                       float* __restrict__ gate) {
  __shared__ float mvec[256];
  __shared__ float hvec[64];
  const int t = threadIdx.x;
  for (int b = 0; b < 16; ++b) {
    mvec[t] = psum[b * 256 + t] * (1.f / 4096.f);
    __syncthreads();
    if (t < 64) {
      float d = f1b[t];
      for (int k = 0; k < 256; ++k) d += f1w[t * 256 + k] * mvec[k];
      hvec[t] = 0.5f * d * (1.f + erff(d * 0.7071067811865475f));  // exact GELU
    }
    __syncthreads();
    float v[3];
#pragma unroll
    for (int j = 0; j < 3; ++j) {
      const int o = t * 3 + j;          // reshape(B,C,3): (b,c,j) = fc2[b, c*3+j]
      float d = f2b[o];
      for (int k = 0; k < 64; ++k) d += f2w[o * 64 + k] * hvec[k];
      v[j] = d;
    }
    const float mx = fmaxf(v[0], fmaxf(v[1], v[2]));
    const float e0 = expf(v[0] - mx), e1 = expf(v[1] - mx), e2 = expf(v[2] - mx);
    const float inv = 1.f / (e0 + e1 + e2);
    gate[0 * 4096 + b * 256 + t] = e0 * inv;
    gate[1 * 4096 + b * 256 + t] = e1 * inv;
    gate[2 * 4096 + b * 256 + t] = e2 * inv;
    __syncthreads();
  }
}

// ---------------------------------------------------------------------------
// Kernel 5: gate-weighted combine (f16 in, 32B vector loads) -> f16
// A-fragment staged in LDS -> WMMA projection GEMM (+ proj bias) -> d_out
// (f32, NHWC).
// ---------------------------------------------------------------------------
__global__ void k_combine_proj(const _Float16* __restrict__ Wb, const _Float16* __restrict__ Hb,
                               const _Float16* __restrict__ Cb, const float* __restrict__ gate,
                               const _Float16* __restrict__ Bp, const float* __restrict__ pbias,
                               float* __restrict__ out) {
  __shared__ _Float16 afrag[8 * 32 * 16];   // 8 KB: one 16x256 f16 A tile, frag order
  const int mtile = blockIdx.x;
  const int bidx  = mtile >> 8;
  const int wave  = threadIdx.x >> 5;
  const int lane  = threadIdx.x & 31;

  {  // stage mixed = g0*w + g1*h + g2*c into LDS in A-fragment layout
    const int m  = threadIdx.x & 15;
    const int c0 = (threadIdx.x >> 4) << 4;        // 16-channel chunk per thread
    const int mg = mtile * 16 + m;
    const size_t rb = (size_t)mg * 256 + c0;       // 32B aligned
    v16h wv = *(const v16h*)(Wb + rb);
    v16h hv = *(const v16h*)(Hb + rb);
    v16h cv = *(const v16h*)(Cb + rb);
#pragma unroll
    for (int j = 0; j < 16; ++j) {
      const int chn = c0 + j;
      const float g0 = gate[0 * 4096 + bidx * 256 + chn];
      const float g1 = gate[1 * 4096 + bidx * 256 + chn];
      const float g2 = gate[2 * 4096 + bidx * 256 + chn];
      const float mix = (float)wv[j] * g0 + (float)hv[j] * g1 + (float)cv[j] * g2;
      const int ks = chn >> 5, kk = chn & 31;
      const int kh = (kk >> 3) & 1;
      const int ln = m | (kh << 4);
      const int e  = ((kk >> 4) << 3) | (kk & 7);
      afrag[((ks * 32 + ln) << 4) + e] = (_Float16)mix;
    }
  }
  __syncthreads();

  const int nt0 = wave * 2;
  v8f acc0 = {}, acc1 = {};
#pragma unroll
  for (int ks = 0; ks < 8; ++ks) {
    v16h a  = *(const v16h*)(afrag + ((ks * 32 + lane) << 4));
    v16h b0 = *(const v16h*)(Bp + (((size_t)((nt0 + 0) * 8 + ks) * 32 + lane) << 4));
    v16h b1 = *(const v16h*)(Bp + (((size_t)((nt0 + 1) * 8 + ks) * 32 + lane) << 4));
    acc0 = __builtin_amdgcn_wmma_f32_16x16x32_f16(false, a, false, b0, (short)0, acc0, false, false);
    acc1 = __builtin_amdgcn_wmma_f32_16x16x32_f16(false, a, false, b1, (short)0, acc1, false, false);
  }

  const int mrow0 = (lane >> 4) << 3;
  const int ncol  = lane & 15;
#pragma unroll
  for (int t = 0; t < 2; ++t) {
    v8f a = t ? acc1 : acc0;
    const int n = (nt0 + t) * 16 + ncol;
    const float pb = pbias[n];
#pragma unroll
    for (int r = 0; r < 8; ++r) {
      const int mg = mtile * 16 + mrow0 + r;
      out[(size_t)mg * 256 + n] = a[r] + pb;
    }
  }
}

// ---------------------------------------------------------------------------
extern "C" void kernel_launch(void* const* d_in, const int* in_sizes, int n_in,
                              void* d_out, int out_size, void* d_ws, size_t ws_size,
                              hipStream_t stream) {
  (void)in_sizes; (void)n_in; (void)out_size; (void)ws_size;
  const float* x   = (const float*)d_in[0];
  const float* off = (const float*)d_in[1];
  const float* wW  = (const float*)d_in[2];
  const float* wB  = (const float*)d_in[3];
  const float* hW  = (const float*)d_in[4];
  const float* hB  = (const float*)d_in[5];
  const float* lw  = (const float*)d_in[6];
  const float* lb  = (const float*)d_in[7];
  const float* f1w = (const float*)d_in[8];
  const float* f1b = (const float*)d_in[9];
  const float* f2w = (const float*)d_in[10];
  const float* f2b = (const float*)d_in[11];
  const float* pW  = (const float*)d_in[12];
  const float* pB  = (const float*)d_in[13];
  float* out = (float*)d_out;

  char* ws = (char*)d_ws;
  _Float16* Bw   = (_Float16*)(ws + 0);                     // 128 KB
  _Float16* Bh   = (_Float16*)(ws + (128 << 10));           // 128 KB
  _Float16* Bp   = (_Float16*)(ws + (256 << 10));           // 128 KB
  float*    gate = (float*)(ws + (384 << 10));              // 48 KB
  float*    psum = (float*)(ws + (448 << 10));              // 16 KB
  _Float16* Aw   = (_Float16*)(ws + (1ull << 20));                         // 32 MB
  _Float16* Ah   = (_Float16*)(ws + (1ull << 20) + (32ull << 20));         // 32 MB
  _Float16* Cb   = (_Float16*)(ws + (1ull << 20) + (64ull << 20));         // 32 MB
  _Float16* Wbuf = (_Float16*)(ws + (1ull << 20) + (96ull << 20));         // 32 MB
  _Float16* Hbuf = (_Float16*)(ws + (1ull << 20) + (128ull << 20));        // 32 MB

  k_zero<<<16, 256, 0, stream>>>(psum, 16 * 256);
  k_prep_b<<<768, 256, 0, stream>>>(wW, hW, pW, Bw, Bh, Bp);
  k_sample<<<4096, 256, 0, stream>>>(x, off, lw, lb, Aw, Ah, Cb);
  k_gemm_wh<<<4096, 256, 0, stream>>>(Aw, Ah, Bw, Bh, wB, hB, Cb, Wbuf, Hbuf, psum);
  k_gate<<<1, 256, 0, stream>>>(psum, f1w, f1b, f2w, f2b, gate);
  k_combine_proj<<<4096, 256, 0, stream>>>(Wbuf, Hbuf, Cb, gate, Bp, pB, out);
}
